// GaussianCompressionModel_32272384262252
// MI455X (gfx1250) — matrix operand order repaired and checked by reading, not compiled, not run on hardware
//
#include <hip/hip_runtime.h>
#include <hip/hip_bf16.h>
#include <stddef.h>

// ---------------------------------------------------------------------------
// VQ nearest-codebook search on gfx1250 (MI455X).
//   argmin_k ||z_n - c_k||^2  ==  argmin_k ( ||c_k||^2 - 2 z_n.c_k )
// The N=65536 x K=8192 x D=64 dot-product matrix runs on
// V_WMMA_F32_16X16X32_BF16 with the fp32 -> bf16 hi+lo split
// (hi*hi + lo*hi + hi*lo ~ fp32 accuracy, 6 WMMAs per 16x16x64 tile).
// Inner loop is ping-pong software-pipelined: tile t+1's eight b128 loads are
// issued before tile t's WMMA chain so the XDL pipe never stalls on loadcnt.
// ---------------------------------------------------------------------------

typedef __attribute__((ext_vector_type(4)))  float  f32x4;
typedef __attribute__((ext_vector_type(8)))  float  v8f;
typedef __attribute__((ext_vector_type(8)))  __bf16 v8bf;
typedef __attribute__((ext_vector_type(16))) __bf16 v16bf;

#define DLAT 64   // latent dim D

// ---- prep 1: split fp32 codebook into bf16 hi + bf16 lo (elementwise) ------
__global__ void vq_prep_split(const float* __restrict__ cb,
                              __bf16* __restrict__ hi,
                              __bf16* __restrict__ lo,
                              int total) {
  int i = blockIdx.x * blockDim.x + threadIdx.x;
  if (i < total) {
    float c = cb[i];
    __bf16 h = (__bf16)c;
    hi[i] = h;
    lo[i] = (__bf16)(c - (float)h);
  }
}

// ---- prep 2: c_sq[k] = ||codebook[k]||^2 -----------------------------------
__global__ void vq_prep_csq(const float* __restrict__ cb,
                            float* __restrict__ csq, int K) {
  int k = blockIdx.x * blockDim.x + threadIdx.x;
  if (k < K) {
    const f32x4* r = (const f32x4*)(cb + (size_t)k * DLAT);
    float s = 0.f;
#pragma unroll
    for (int i = 0; i < DLAT / 4; ++i) {
      f32x4 p = r[i];
      s = fmaf(p[0], p[0], s);
      s = fmaf(p[1], p[1], s);
      s = fmaf(p[2], p[2], s);
      s = fmaf(p[3], p[3], s);
    }
    csq[k] = s;
  }
}

// ---- B-tile fragment load: 8 x b128, constant immediate offsets ------------
__device__ __forceinline__ void vq_loadB(const __bf16* __restrict__ ph,
                                         const __bf16* __restrict__ pl,
                                         v8bf h[4], v8bf l[4]) {
  h[0] = *(const v8bf*)(ph + 0);
  h[1] = *(const v8bf*)(ph + 8);
  h[2] = *(const v8bf*)(ph + 32);
  h[3] = *(const v8bf*)(ph + 40);
  l[0] = *(const v8bf*)(pl + 0);
  l[1] = *(const v8bf*)(pl + 8);
  l[2] = *(const v8bf*)(pl + 32);
  l[3] = *(const v8bf*)(pl + 40);
}

// ---- one 16-column tile: 6 WMMAs + argmin update ---------------------------
__device__ __forceinline__ void vq_tile(const v8bf h[4], const v8bf l[4],
                                        const v16bf a_hi[2], const v16bf a_lo[2],
                                        float cs, int kc,
                                        float best_s[8], int best_i[8]) {
  v16bf Bh0 = __builtin_shufflevector(h[0], h[1], 0, 1, 2, 3, 4, 5, 6, 7,
                                      8, 9, 10, 11, 12, 13, 14, 15);
  v16bf Bl0 = __builtin_shufflevector(l[0], l[1], 0, 1, 2, 3, 4, 5, 6, 7,
                                      8, 9, 10, 11, 12, 13, 14, 15);
  v16bf Bh1 = __builtin_shufflevector(h[2], h[3], 0, 1, 2, 3, 4, 5, 6, 7,
                                      8, 9, 10, 11, 12, 13, 14, 15);
  v16bf Bl1 = __builtin_shufflevector(l[2], l[3], 0, 1, 2, 3, 4, 5, 6, 7,
                                      8, 9, 10, 11, 12, 13, 14, 15);
  v8f acc = {0.f, 0.f, 0.f, 0.f, 0.f, 0.f, 0.f, 0.f};
  // hi*hi + lo*hi + hi*lo per 32-wide D chunk (lo*lo below fp32 ulp)
  acc = __builtin_amdgcn_wmma_f32_16x16x32_bf16(false, a_hi[0], false, Bh0,
                                                (short)0, acc, false, false);
  acc = __builtin_amdgcn_wmma_f32_16x16x32_bf16(false, a_lo[0], false, Bh0,
                                                (short)0, acc, false, false);
  acc = __builtin_amdgcn_wmma_f32_16x16x32_bf16(false, a_hi[0], false, Bl0,
                                                (short)0, acc, false, false);
  acc = __builtin_amdgcn_wmma_f32_16x16x32_bf16(false, a_hi[1], false, Bh1,
                                                (short)0, acc, false, false);
  acc = __builtin_amdgcn_wmma_f32_16x16x32_bf16(false, a_lo[1], false, Bh1,
                                                (short)0, acc, false, false);
  acc = __builtin_amdgcn_wmma_f32_16x16x32_bf16(false, a_hi[1], false, Bl1,
                                                (short)0, acc, false, false);
#pragma unroll
  for (int v = 0; v < 8; ++v) {
    float s = fmaf(-2.0f, acc[v], cs);
    if (s < best_s[v]) { best_s[v] = s; best_i[v] = kc; }  // strict < keeps first k
  }
}

// ---- main kernel -----------------------------------------------------------
// 256 threads = 8 wave32; wave w handles rows [m0, m0+16), block covers 128
// rows. Each wave sweeps all K in 16-column tiles.
__global__ __launch_bounds__(256) void vq_argmin_wmma(
    const float*  __restrict__ z,        // [N,64] fp32
    const float*  __restrict__ codebook, // [K,64] fp32 (for z_q gather)
    const __bf16* __restrict__ bhi,      // [K,64] bf16 hi
    const __bf16* __restrict__ blo,      // [K,64] bf16 lo
    const float*  __restrict__ csq,      // [K]
    float* __restrict__ zq,              // [N,64]
    float* __restrict__ idx_out,         // [N] (indices as float)
    int K) {
  const int lane    = threadIdx.x & 31;
  const int wave    = threadIdx.x >> 5;
  const int m0      = blockIdx.x * 128 + wave * 16;
  const int sub     = lane & 15;  // lane-within-half
  const int halfsel = lane >> 4;  // 0: lanes 0-15, 1: lanes 16-31

  // ---- Wave-invariant A fragments (16-bit A 16x32, this code's contiguous k order, not the ISA table order):
  //   lanes 0-15  hold row M=lane,    inner-K {0..15}
  //   lanes 16-31 hold row M=lane-16, inner-K {16..31}
  v16bf a_hi[2], a_lo[2];
  {
    const float* zr = z + (size_t)(m0 + sub) * DLAT;
#pragma unroll
    for (int c = 0; c < 2; ++c) {
      const int g1 = 32 * c + 16 * halfsel;
      const int g2 = g1 + 8;
      f32x4 p0 = *(const f32x4*)(zr + g1);
      f32x4 p1 = *(const f32x4*)(zr + g1 + 4);
      f32x4 p2 = *(const f32x4*)(zr + g2);
      f32x4 p3 = *(const f32x4*)(zr + g2 + 4);
      float t[16];
#pragma unroll
      for (int e = 0; e < 4; ++e) {
        t[e] = p0[e]; t[4 + e] = p1[e]; t[8 + e] = p2[e]; t[12 + e] = p3[e];
      }
#pragma unroll
      for (int e = 0; e < 16; ++e) {
        __bf16 h = (__bf16)t[e];
        a_hi[c][e] = h;
        a_lo[c][e] = (__bf16)(t[e] - (float)h);
      }
    }
  }

  float best_s[8];
  int   best_i[8];
#pragma unroll
  for (int v = 0; v < 8; ++v) { best_s[v] = 3.4e38f; best_i[v] = 0; }

  // Per-lane B base (column = k-tile + sub; B layout: lanes0-15 hold inner-K
  // 0-15, lanes16-31 hold inner-K 16-31). All tile loads use immediate
  // offsets off this pointer; it advances by a uniform 2 tiles (+4KB) per
  // loop iteration.
  const int dstartB = 16 * halfsel;
  const __bf16* ph = bhi + (size_t)sub * DLAT + dstartB;
  const __bf16* pl = blo + (size_t)sub * DLAT + dstartB;
  const float*  pc = csq + sub;

  const int T = K / 16;          // 512 tiles (even)
  v8bf h0[4], l0[4], h1[4], l1[4];
  vq_loadB(ph, pl, h0, l0);      // prologue: tile 0
  float cs0 = pc[0];
  int kc = sub;

  // Steady state: process tiles t and t+1, preloading t+1 and t+2 ahead of
  // their WMMA chains (ping-pong buffers -> partial loadcnt waits only).
  for (int t = 0; t < T - 2; t += 2) {
    __builtin_prefetch(ph + 8 * 1024);          // +8 tiles -> global_prefetch_b8
    __builtin_prefetch(pl + 8 * 1024);
    vq_loadB(ph + 1024, pl + 1024, h1, l1);     // tile t+1
    float cs1 = pc[16];
    vq_tile(h0, l0, a_hi, a_lo, cs0, kc, best_s, best_i);
    vq_loadB(ph + 2048, pl + 2048, h0, l0);     // tile t+2
    cs0 = pc[32];
    vq_tile(h1, l1, a_hi, a_lo, cs1, kc + 16, best_s, best_i);
    ph += 2048; pl += 2048; pc += 32; kc += 32;
  }
  // Epilogue: tiles T-2, T-1
  vq_loadB(ph + 1024, pl + 1024, h1, l1);
  float cs1 = pc[16];
  vq_tile(h0, l0, a_hi, a_lo, cs0, kc, best_s, best_i);
  vq_tile(h1, l1, a_hi, a_lo, cs1, kc + 16, best_s, best_i);

  // ---- Cross-lane argmin reduction within each 16-lane half (xor masks
  // 1,2,4,8 stay inside a half). Tie-break on smaller index = first match.
#pragma unroll
  for (int m = 1; m <= 8; m <<= 1) {
#pragma unroll
    for (int v = 0; v < 8; ++v) {
      float os = __shfl_xor(best_s[v], m, 32);
      int   oi = __shfl_xor(best_i[v], m, 32);
      if (os < best_s[v] || (os == best_s[v] && oi < best_i[v])) {
        best_s[v] = os;
        best_i[v] = oi;
      }
    }
  }

  // ---- Output: row of VGPR v in this half = m0 + v + 8*halfsel. The 16
  // lanes of the half gather the winning fp32 codebook row cooperatively.
#pragma unroll
  for (int v = 0; v < 8; ++v) {
    const int row = m0 + v + 8 * halfsel;
    const int idx = best_i[v];
    const f32x4* src = (const f32x4*)(codebook + (size_t)idx * DLAT);
    ((f32x4*)(zq + (size_t)row * DLAT))[sub] = src[sub];
    if (sub == 0) idx_out[row] = (float)idx;
  }
}

extern "C" void kernel_launch(void* const* d_in, const int* in_sizes, int n_in,
                              void* d_out, int out_size, void* d_ws, size_t ws_size,
                              hipStream_t stream) {
  const float* z  = (const float*)d_in[0];  // [N,64]
  const float* cb = (const float*)d_in[1];  // [K,64]
  const int N = in_sizes[0] / DLAT;  // 65536
  const int K = in_sizes[1] / DLAT;  // 8192

  // workspace: bf16 hi [K*64] | bf16 lo [K*64] | float csq [K]  (~2.03 MB)
  __bf16* hi  = (__bf16*)d_ws;
  __bf16* lo  = hi + (size_t)K * DLAT;
  float*  csq = (float*)(lo + (size_t)K * DLAT);

  float* zq      = (float*)d_out;               // [N,64]
  float* idx_out = zq + (size_t)N * DLAT;       // [N]

  const int total = K * DLAT;
  vq_prep_split<<<(total + 255) / 256, 256, 0, stream>>>(cb, hi, lo, total);
  vq_prep_csq<<<(K + 255) / 256, 256, 0, stream>>>(cb, csq, K);
  vq_argmin_wmma<<<N / 128, 256, 0, stream>>>(z, cb, hi, lo, csq, zq, idx_out, K);
}
